// InvNet_3178275799542
// MI455X (gfx1250) — matrix operand order repaired and checked by reading, not compiled, not run on hardware
//
#include <hip/hip_runtime.h>
#include <hip/hip_bf16.h>
#include <stdint.h>

// ---------------------------------------------------------------------------
// MI455X (gfx1250): split-bf16 WMMA GEMMs (x ~= hi+lo; x*y ~= hihi+hilo+lohi),
// em pre-converted once to bf16 hi/lo planes (kills redundant per-tile cvt
// VALU work), tiles streamed into LDS with GLOBAL_LOAD_ASYNC_TO_LDS_B128/B64
// double-buffered behind s_wait_asynccnt. Runtime fallback if ws too small.
// ---------------------------------------------------------------------------

typedef __bf16 bf16;
typedef __attribute__((ext_vector_type(16))) __bf16 v16bf;
typedef __attribute__((ext_vector_type(8)))  float  v8f;

// async-to-LDS builtin pointer types (from hipcc diagnostic: param is
// "vector_size(16) int __device__*" / "vector_size(8) int __device__*")
typedef int i4v __attribute__((vector_size(16)));
typedef int i2v __attribute__((vector_size(8)));
typedef __attribute__((address_space(1))) i4v as1_i4;
typedef __attribute__((address_space(3))) i4v as3_i4;
typedef __attribute__((address_space(1))) i2v as1_i2;
typedef __attribute__((address_space(3))) i2v as3_i2;

#define DIM     2048
#define NEM     16384
#define BROWS   256
#define KNN6    6
#define INV_BETA 20.0f     // 1/0.05
#define NSPLIT  8
#define NCHUNK  (NEM / NSPLIT)   // 2048 columns per sims block
#define KSLABS  (DIM / 32)       // 64
#define NEG_BIG (-3.402823466e+38f)
#define EMTOT   ((size_t)NEM * DIM)     // 33,554,432
#define INTOT   ((size_t)BROWS * DIM)   // 524,288

#ifndef __has_builtin
#define __has_builtin(x) 0
#endif
#if defined(__gfx1250__) && \
    __has_builtin(__builtin_amdgcn_global_load_async_to_lds_b128) && \
    __has_builtin(__builtin_amdgcn_global_load_async_to_lds_b64) && \
    __has_builtin(__builtin_amdgcn_s_wait_asynccnt)
#define HAVE_ASYNC 1
#else
#define HAVE_ASYNC 0
#endif

#if HAVE_ASYNC
#define ASYNC_WAIT(n) __builtin_amdgcn_s_wait_asynccnt(n)
#else
#define ASYNC_WAIT(n) do {} while (0)
#endif

// ---- async (or sync-fallback) global -> LDS copies -------------------------
__device__ __forceinline__ void cp16_g2l(bf16* dst_lds, const bf16* src_g) {
#if HAVE_ASYNC
  __builtin_amdgcn_global_load_async_to_lds_b128((as1_i4*)src_g,
                                                 (as3_i4*)dst_lds, 0, 0);
#else
  *(uint4*)dst_lds = *(const uint4*)src_g;
#endif
}
__device__ __forceinline__ void cp8_g2l(bf16* dst_lds, const bf16* src_g) {
#if HAVE_ASYNC
  __builtin_amdgcn_global_load_async_to_lds_b64((as1_i2*)src_g,
                                                (as3_i2*)dst_lds, 0, 0);
#else
  *(uint2*)dst_lds = *(const uint2*)src_g;
#endif
}

// ---- bf16 helpers (bit-exact RNE) ------------------------------------------
__device__ __forceinline__ bf16 f2bf(float x) {
  unsigned u = __float_as_uint(x);
  unsigned r = (u + 0x7fffu + ((u >> 16) & 1u)) >> 16;
  unsigned short hs = (unsigned short)r;
  bf16 b; __builtin_memcpy(&b, &hs, 2);
  return b;
}
__device__ __forceinline__ float bf2f(bf16 b) {
  unsigned short hs; __builtin_memcpy(&hs, &b, 2);
  return __uint_as_float(((unsigned)hs) << 16);
}
__device__ __forceinline__ void split2(float x, bf16* hi, bf16* lo) {
  bf16 h = f2bf(x);
  *hi = h;
  *lo = f2bf(x - bf2f(h));
}

// ---- WMMA wrapper -----------------------------------------------------------
__device__ __forceinline__ v8f wmma_bf16(v16bf a, v16bf b, v8f c) {
  return __builtin_amdgcn_wmma_f32_16x16x32_bf16(
      false, a, false, b, (short)0, c, false, false);
}

// A fragment from LDS tile [16 rows][32 K] bf16 (two 16B chunks per lane).
__device__ __forceinline__ v16bf frag_a(const bf16* base) {
  const int l = threadIdx.x & 31;
  const int h = l >> 4, m = l & 15;
  v16bf f;
  ((uint4*)&f)[0] = *(const uint4*)(base + m * 32 + 8 * h);
  ((uint4*)&f)[1] = *(const uint4*)(base + m * 32 + 16 + 8 * h);
  return f;
}
// B fragment from LDS tile [16 cols][32 K] bf16 (two 16B chunks per lane).
__device__ __forceinline__ v16bf frag_b(const bf16* base) {
  const int l = threadIdx.x & 31;
  const int h = l >> 4, n = l & 15;
  v16bf f;
  ((uint4*)&f)[0] = *(const uint4*)(base + n * 32 + 8 * h);
  ((uint4*)&f)[1] = *(const uint4*)(base + n * 32 + 16 + 8 * h);
  return f;
}

// ---- top-6 sorted insert (descending value, ties -> lower index) -----------
__device__ __forceinline__ void top6_ins(float* v, int* ix, float nv, int ni) {
#pragma unroll
  for (int p = 0; p < 6; p++) {
    if (nv > v[p] || (nv == v[p] && ni < ix[p])) {
#pragma unroll
      for (int q = 5; q > p; q--) { v[q] = v[q - 1]; ix[q] = ix[q - 1]; }
      v[p] = nv; ix[p] = ni;
      return;
    }
  }
}

// ===========================================================================
// Kernel 0: one-time fp32 -> split-bf16 plane conversion (em and inputs).
// ===========================================================================
__global__ __launch_bounds__(256, 1)
void k_convert(const float* __restrict__ em, const float* __restrict__ inp,
               bf16* __restrict__ emhi, bf16* __restrict__ emlo,
               bf16* __restrict__ inphi, bf16* __restrict__ inplo) {
  const size_t i = ((size_t)blockIdx.x * 256 + threadIdx.x) * 8;
  const float* src; bf16 *dhi, *dlo;
  if (i < EMTOT) {
    src = em + i; dhi = emhi + i; dlo = emlo + i;
  } else {
    const size_t j = i - EMTOT;
    if (j >= INTOT) return;
    src = inp + j; dhi = inphi + j; dlo = inplo + j;
  }
  float4 f0 = ((const float4*)src)[0];
  float4 f1 = ((const float4*)src)[1];
  float vv[8] = {f0.x, f0.y, f0.z, f0.w, f1.x, f1.y, f1.z, f1.w};
  bf16 hb[8], lb[8];
#pragma unroll
  for (int q = 0; q < 8; q++) split2(vv[q], &hb[q], &lb[q]);
  *(uint4*)dhi = *(uint4*)hb;
  *(uint4*)dlo = *(uint4*)lb;
}

// ===========================================================================
// Kernel 1 (plane path): logits = (inputs @ em^T)*20, async double-buffered.
// Block 256 thr (8 waves, 2x4), tile 64M x 128N, K-slab 32.
// ===========================================================================
__global__ __launch_bounds__(256, 1)
void k_logits_p(const bf16* __restrict__ Ahi, const bf16* __restrict__ Alo,
                const bf16* __restrict__ Bhi, const bf16* __restrict__ Blo,
                float* __restrict__ logits) {
  __shared__ bf16 sA[2][2][64 * 32];    // [buf][hi/lo]
  __shared__ bf16 sB[2][2][128 * 32];
  const int t = threadIdx.x;
  const int n0 = blockIdx.x * 128, m0 = blockIdx.y * 64;
  const int w = t >> 5, wm = w & 1, wn = w >> 1;

  v8f acc[2][2];
#pragma unroll
  for (int i = 0; i < 2; i++)
#pragma unroll
    for (int j = 0; j < 2; j++) acc[i][j] = (v8f){0, 0, 0, 0, 0, 0, 0, 0};

  auto load_slab = [&](int ks, int buf) {  // 6 copies per thread (uniform)
    const int k0 = ks * 32;
#pragma unroll
    for (int q = 0; q < 2; q++) {  // A: 512 x 16B chunks
      const int c = t + q * 256;
      const int pl = c >> 8, r = (c & 255) >> 2, off = (c & 3) * 8;
      const bf16* g = (pl ? Alo : Ahi) + (size_t)(m0 + r) * DIM + k0 + off;
      cp16_g2l(&sA[buf][pl][r * 32 + off], g);
    }
#pragma unroll
    for (int q = 0; q < 4; q++) {  // B: 1024 x 16B chunks
      const int c = t + q * 256;
      const int pl = c >> 9, r = (c & 511) >> 2, off = (c & 3) * 8;
      const bf16* g = (pl ? Blo : Bhi) + (size_t)(n0 + r) * DIM + k0 + off;
      cp16_g2l(&sB[buf][pl][r * 32 + off], g);
    }
  };

  load_slab(0, 0);
  for (int ks = 0; ks < KSLABS; ks++) {
    const int buf = ks & 1;
    if (ks + 1 < KSLABS) { load_slab(ks + 1, buf ^ 1); ASYNC_WAIT(6); }
    else                 { ASYNC_WAIT(0); }
    __syncthreads();   // slab ks visible to all; all issues done

    v16bf ah[2], al[2], bh[2], bl[2];
#pragma unroll
    for (int i = 0; i < 2; i++) {
      ah[i] = frag_a(&sA[buf][0][(wm * 32 + i * 16) * 32]);
      al[i] = frag_a(&sA[buf][1][(wm * 32 + i * 16) * 32]);
      bh[i] = frag_b(&sB[buf][0][(wn * 32 + i * 16) * 32]);
      bl[i] = frag_b(&sB[buf][1][(wn * 32 + i * 16) * 32]);
    }
#pragma unroll
    for (int i = 0; i < 2; i++)
#pragma unroll
      for (int j = 0; j < 2; j++) {
        acc[i][j] = wmma_bf16(al[i], bh[j], acc[i][j]);
        acc[i][j] = wmma_bf16(ah[i], bl[j], acc[i][j]);
        acc[i][j] = wmma_bf16(ah[i], bh[j], acc[i][j]);
      }
    __syncthreads();   // all reads of buf done before it is refilled
  }

  const int l = t & 31, hh = l >> 4, nn = l & 15;
#pragma unroll
  for (int i = 0; i < 2; i++)
#pragma unroll
    for (int j = 0; j < 2; j++)
#pragma unroll
      for (int r = 0; r < 8; r++) {
        const int mm = m0 + wm * 32 + i * 16 + r + 8 * hh;
        const int cc = n0 + wn * 32 + j * 16 + nn;
        logits[(size_t)mm * NEM + cc] = acc[i][j][r] * INV_BETA;
      }
}

// ===========================================================================
// Kernel 1 (fallback): on-the-fly split conversion (no planes in ws).
// ===========================================================================
__global__ __launch_bounds__(256, 1)
void k_logits_f(const float* __restrict__ inp, const float* __restrict__ em,
                float* __restrict__ logits) {
  __shared__ bf16 sAhi[64 * 32], sAlo[64 * 32];
  __shared__ bf16 sBhi[128 * 32], sBlo[128 * 32];
  const int t = threadIdx.x;
  const int n0 = blockIdx.x * 128, m0 = blockIdx.y * 64;
  const int w = t >> 5, wm = w & 1, wn = w >> 1;

  v8f acc[2][2];
#pragma unroll
  for (int i = 0; i < 2; i++)
#pragma unroll
    for (int j = 0; j < 2; j++) acc[i][j] = (v8f){0, 0, 0, 0, 0, 0, 0, 0};

  for (int k0 = 0; k0 < DIM; k0 += 32) {
    __syncthreads();
    {
      const int r = t >> 2, c = (t & 3) << 3;
      const float* g = inp + (size_t)(m0 + r) * DIM + k0 + c;
      float4 f0 = ((const float4*)g)[0];
      float4 f1 = ((const float4*)g)[1];
      float vv[8] = {f0.x, f0.y, f0.z, f0.w, f1.x, f1.y, f1.z, f1.w};
#pragma unroll
      for (int i = 0; i < 8; i++)
        split2(vv[i], &sAhi[r * 32 + c + i], &sAlo[r * 32 + c + i]);
    }
    {
      const int r = t >> 1, c = (t & 1) << 4;
      const float* g = em + (size_t)(n0 + r) * DIM + k0 + c;
#pragma unroll
      for (int q = 0; q < 4; q++) {
        float4 f = ((const float4*)g)[q];
        float vv[4] = {f.x, f.y, f.z, f.w};
#pragma unroll
        for (int i = 0; i < 4; i++)
          split2(vv[i], &sBhi[r * 32 + c + q * 4 + i],
                        &sBlo[r * 32 + c + q * 4 + i]);
      }
      if (k0 + 32 < DIM) __builtin_prefetch(g + 32, 0, 1);
    }
    __syncthreads();

    v16bf ah[2], al[2], bh[2], bl[2];
#pragma unroll
    for (int i = 0; i < 2; i++) {
      ah[i] = frag_a(sAhi + (wm * 32 + i * 16) * 32);
      al[i] = frag_a(sAlo + (wm * 32 + i * 16) * 32);
      bh[i] = frag_b(sBhi + (wn * 32 + i * 16) * 32);
      bl[i] = frag_b(sBlo + (wn * 32 + i * 16) * 32);
    }
#pragma unroll
    for (int i = 0; i < 2; i++)
#pragma unroll
      for (int j = 0; j < 2; j++) {
        acc[i][j] = wmma_bf16(al[i], bh[j], acc[i][j]);
        acc[i][j] = wmma_bf16(ah[i], bl[j], acc[i][j]);
        acc[i][j] = wmma_bf16(ah[i], bh[j], acc[i][j]);
      }
  }
  const int l = t & 31, hh = l >> 4, nn = l & 15;
#pragma unroll
  for (int i = 0; i < 2; i++)
#pragma unroll
    for (int j = 0; j < 2; j++)
#pragma unroll
      for (int r = 0; r < 8; r++) {
        const int mm = m0 + wm * 32 + i * 16 + r + 8 * hh;
        const int cc = n0 + wn * 32 + j * 16 + nn;
        logits[(size_t)mm * NEM + cc] = acc[i][j][r] * INV_BETA;
      }
}

// ===========================================================================
// Kernel 2: per-row max, sum(exp(x-max)), top-6 of logits. One block/row.
// ===========================================================================
__global__ __launch_bounds__(256, 1)
void k_rowstats(const float* __restrict__ logits, float* __restrict__ t6v,
                int* __restrict__ t6i, float* __restrict__ rowmax,
                float* __restrict__ rowsum) {
  __shared__ float s1v[256 * 6];
  __shared__ int   s1i[256 * 6];
  __shared__ float s2v[16 * 6];
  __shared__ int   s2i[16 * 6];
  __shared__ float sred[256];
  __shared__ float sMax;
  const int m = blockIdx.x, t = threadIdx.x;
  const float* row = logits + (size_t)m * NEM;

  float tv[6]; int ti[6];
#pragma unroll
  for (int i = 0; i < 6; i++) { tv[i] = NEG_BIG; ti[i] = 0x7fffffff; }
  for (int c = t; c < NEM; c += 256) top6_ins(tv, ti, row[c], c);
#pragma unroll
  for (int i = 0; i < 6; i++) { s1v[t * 6 + i] = tv[i]; s1i[t * 6 + i] = ti[i]; }
  __syncthreads();

  if (t < 16) {
    float mv[6]; int mi[6];
#pragma unroll
    for (int i = 0; i < 6; i++) { mv[i] = NEG_BIG; mi[i] = 0x7fffffff; }
    for (int s = 0; s < 16; s++)
      for (int i = 0; i < 6; i++)
        top6_ins(mv, mi, s1v[(t * 16 + s) * 6 + i], s1i[(t * 16 + s) * 6 + i]);
#pragma unroll
    for (int i = 0; i < 6; i++) { s2v[t * 6 + i] = mv[i]; s2i[t * 6 + i] = mi[i]; }
  }
  __syncthreads();

  if (t == 0) {
    float mv[6]; int mi[6];
#pragma unroll
    for (int i = 0; i < 6; i++) { mv[i] = NEG_BIG; mi[i] = 0x7fffffff; }
    for (int s = 0; s < 16; s++)
      for (int i = 0; i < 6; i++)
        top6_ins(mv, mi, s2v[s * 6 + i], s2i[s * 6 + i]);
#pragma unroll
    for (int i = 0; i < 6; i++) { t6v[m * 6 + i] = mv[i]; t6i[m * 6 + i] = mi[i]; }
    rowmax[m] = mv[0];
    sMax = mv[0];
  }
  __syncthreads();

  const float mx = sMax;
  float acc = 0.f;
  for (int c = t; c < NEM; c += 256) acc += expf(row[c] - mx);
  sred[t] = acc;
  __syncthreads();
  for (int off = 128; off > 0; off >>= 1) {
    if (t < off) sred[t] += sred[t + off];
    __syncthreads();
  }
  if (t == 0) rowsum[m] = sred[0];
}

// ===========================================================================
// Kernel 3 (plane path): sims = em[top6] @ em^T, async double-buffered,
// fused online top-6. grid = (96 row-groups, 8 N-splits).
// ===========================================================================
__global__ __launch_bounds__(256, 1)
void k_sims_p(const bf16* __restrict__ Ehi, const bf16* __restrict__ Elo,
              const int* __restrict__ t6i, float* __restrict__ pt6v,
              int* __restrict__ pt6i) {
  __shared__ bf16 sA[2][2][16 * 32];
  __shared__ bf16 sB[2][2][128 * 32];
  __shared__ float sC[16 * 128];
  __shared__ int   sGi[16];
  __shared__ float s1v[256 * 6];
  __shared__ int   s1i[256 * 6];

  const int t = threadIdx.x, w = t >> 5;
  const int grp = blockIdx.x, sp = blockIdx.y;
  if (t < 16) {
    const int g = grp * 16 + t;
    sGi[t] = t6i[(g / 6) * 6 + (g % 6)];
  }
  __syncthreads();

  float tv[6]; int ti[6];
#pragma unroll
  for (int i = 0; i < 6; i++) { tv[i] = NEG_BIG; ti[i] = 0x7fffffff; }
  const int rr = t >> 4, cc = t & 15;

  // per-thread A b64 chunk: 2 planes x 128 x 8B
  const int apl = t >> 7, aidx = t & 127, ar = aidx >> 3, aoff = (aidx & 7) * 4;
  const size_t arowbase = (size_t)sGi[ar] * DIM;

  for (int nt = 0; nt < NCHUNK / 128; nt++) {
    const int n0 = sp * NCHUNK + nt * 128;
    v8f acc = (v8f){0, 0, 0, 0, 0, 0, 0, 0};

    auto load_slab = [&](int ks, int buf) {  // 5 copies per thread (uniform)
      const int k0 = ks * 32;
      cp8_g2l(&sA[buf][apl][ar * 32 + aoff],
              (apl ? Elo : Ehi) + arowbase + k0 + aoff);
#pragma unroll
      for (int q = 0; q < 4; q++) {  // B: 1024 x 16B chunks
        const int c = t + q * 256;
        const int pl = c >> 9, r = (c & 511) >> 2, off = (c & 3) * 8;
        const bf16* g = (pl ? Elo : Ehi) + (size_t)(n0 + r) * DIM + k0 + off;
        cp16_g2l(&sB[buf][pl][r * 32 + off], g);
      }
    };

    load_slab(0, 0);
    for (int ks = 0; ks < KSLABS; ks++) {
      const int buf = ks & 1;
      if (ks + 1 < KSLABS) { load_slab(ks + 1, buf ^ 1); ASYNC_WAIT(5); }
      else                 { ASYNC_WAIT(0); }
      __syncthreads();
      const v16bf ah = frag_a(&sA[buf][0][0]);
      const v16bf al = frag_a(&sA[buf][1][0]);
      const v16bf bh = frag_b(&sB[buf][0][w * 16 * 32]);
      const v16bf bl = frag_b(&sB[buf][1][w * 16 * 32]);
      acc = wmma_bf16(al, bh, acc);
      acc = wmma_bf16(ah, bl, acc);
      acc = wmma_bf16(ah, bh, acc);
      __syncthreads();
    }

    {
      const int l = t & 31, hh = l >> 4, nn2 = l & 15;
#pragma unroll
      for (int r = 0; r < 8; r++)
        sC[(r + 8 * hh) * 128 + w * 16 + nn2] = acc[r];
    }
    __syncthreads();
#pragma unroll
    for (int j = 0; j < 8; j++) {
      const float v = sC[rr * 128 + cc + 16 * j];
      top6_ins(tv, ti, v, n0 + cc + 16 * j);
    }
    __syncthreads();
  }

#pragma unroll
  for (int i = 0; i < 6; i++) { s1v[t * 6 + i] = tv[i]; s1i[t * 6 + i] = ti[i]; }
  __syncthreads();
  if (t < 16) {
    float mv[6]; int mi[6];
#pragma unroll
    for (int i = 0; i < 6; i++) { mv[i] = NEG_BIG; mi[i] = 0x7fffffff; }
    for (int s = 0; s < 16; s++)
      for (int i = 0; i < 6; i++)
        top6_ins(mv, mi, s1v[(t * 16 + s) * 6 + i], s1i[(t * 16 + s) * 6 + i]);
    const int g = grp * 16 + t;
#pragma unroll
    for (int i = 0; i < 6; i++) {
      pt6v[(g * NSPLIT + sp) * 6 + i] = mv[i];
      pt6i[(g * NSPLIT + sp) * 6 + i] = mi[i];
    }
  }
}

// ===========================================================================
// Kernel 3 (fallback): on-the-fly conversion version.
// ===========================================================================
__global__ __launch_bounds__(256, 1)
void k_sims_f(const float* __restrict__ em, const int* __restrict__ t6i,
              float* __restrict__ pt6v, int* __restrict__ pt6i) {
  __shared__ bf16 sAhi[16 * 32], sAlo[16 * 32];
  __shared__ bf16 sBhi[128 * 32], sBlo[128 * 32];
  __shared__ float sC[16 * 128];
  __shared__ int   sGi[16];
  __shared__ float s1v[256 * 6];
  __shared__ int   s1i[256 * 6];

  const int t = threadIdx.x, w = t >> 5;
  const int grp = blockIdx.x, sp = blockIdx.y;
  if (t < 16) {
    const int g = grp * 16 + t;
    sGi[t] = t6i[(g / 6) * 6 + (g % 6)];
  }
  __syncthreads();

  float tv[6]; int ti[6];
#pragma unroll
  for (int i = 0; i < 6; i++) { tv[i] = NEG_BIG; ti[i] = 0x7fffffff; }
  const int rr = t >> 4, cc = t & 15;

  for (int nt = 0; nt < NCHUNK / 128; nt++) {
    const int n0 = sp * NCHUNK + nt * 128;
    v8f acc = (v8f){0, 0, 0, 0, 0, 0, 0, 0};

    for (int k0 = 0; k0 < DIM; k0 += 32) {
      __syncthreads();
      {
        const int r = t >> 4, c = (t & 15) << 1;
        const float* g = em + (size_t)sGi[r] * DIM + k0 + c;
        float2 f = *(const float2*)g;
        split2(f.x, &sAhi[r * 32 + c],     &sAlo[r * 32 + c]);
        split2(f.y, &sAhi[r * 32 + c + 1], &sAlo[r * 32 + c + 1]);
      }
      {
        const int r = t >> 1, c = (t & 1) << 4;
        const float* g = em + (size_t)(n0 + r) * DIM + k0 + c;
#pragma unroll
        for (int q = 0; q < 4; q++) {
          float4 f = ((const float4*)g)[q];
          float vv[4] = {f.x, f.y, f.z, f.w};
#pragma unroll
          for (int i = 0; i < 4; i++)
            split2(vv[i], &sBhi[r * 32 + c + q * 4 + i],
                          &sBlo[r * 32 + c + q * 4 + i]);
        }
        if (k0 + 32 < DIM) __builtin_prefetch(g + 32, 0, 1);
      }
      __syncthreads();

      const v16bf ah = frag_a(sAhi), al = frag_a(sAlo);
      const v16bf bh = frag_b(sBhi + w * 16 * 32);
      const v16bf bl = frag_b(sBlo + w * 16 * 32);
      acc = wmma_bf16(al, bh, acc);
      acc = wmma_bf16(ah, bl, acc);
      acc = wmma_bf16(ah, bh, acc);
    }

    __syncthreads();
    {
      const int l = t & 31, hh = l >> 4, nn2 = l & 15;
#pragma unroll
      for (int r = 0; r < 8; r++)
        sC[(r + 8 * hh) * 128 + w * 16 + nn2] = acc[r];
    }
    __syncthreads();
#pragma unroll
    for (int j = 0; j < 8; j++) {
      const float v = sC[rr * 128 + cc + 16 * j];
      top6_ins(tv, ti, v, n0 + cc + 16 * j);
    }
  }

#pragma unroll
  for (int i = 0; i < 6; i++) { s1v[t * 6 + i] = tv[i]; s1i[t * 6 + i] = ti[i]; }
  __syncthreads();
  if (t < 16) {
    float mv[6]; int mi[6];
#pragma unroll
    for (int i = 0; i < 6; i++) { mv[i] = NEG_BIG; mi[i] = 0x7fffffff; }
    for (int s = 0; s < 16; s++)
      for (int i = 0; i < 6; i++)
        top6_ins(mv, mi, s1v[(t * 16 + s) * 6 + i], s1i[(t * 16 + s) * 6 + i]);
    const int g = grp * 16 + t;
#pragma unroll
    for (int i = 0; i < 6; i++) {
      pt6v[(g * NSPLIT + sp) * 6 + i] = mv[i];
      pt6i[(g * NSPLIT + sp) * 6 + i] = mi[i];
    }
  }
}

// ===========================================================================
// Kernel 4: merge N-split partial top-6 per sims row, anchor membership.
// ===========================================================================
__global__ __launch_bounds__(256, 1)
void k_recip(const float* __restrict__ pt6v, const int* __restrict__ pt6i,
             const int* __restrict__ t6i, int* __restrict__ recip) {
  const int g = blockIdx.x * 256 + threadIdx.x;
  if (g >= BROWS * KNN6) return;
  float mv[6]; int mi[6];
#pragma unroll
  for (int i = 0; i < 6; i++) { mv[i] = NEG_BIG; mi[i] = 0x7fffffff; }
  for (int s = 0; s < NSPLIT; s++)
    for (int i = 0; i < 6; i++)
      top6_ins(mv, mi, pt6v[(g * NSPLIT + s) * 6 + i],
                        pt6i[(g * NSPLIT + s) * 6 + i]);
  const int anchor = t6i[(g / 6) * 6];
  int f = 0;
#pragma unroll
  for (int i = 0; i < 6; i++) f |= (mi[i] == anchor);
  recip[g] = f;
}

// ===========================================================================
// Kernel 5: losses from <=7 special columns per row (softmax sums to 1).
// ===========================================================================
__global__ __launch_bounds__(256, 1)
void k_loss(const float* __restrict__ logits, const float* __restrict__ t6v,
            const int* __restrict__ t6i, const float* __restrict__ rowmax,
            const float* __restrict__ rowsum, const int* __restrict__ recip,
            const long long* __restrict__ targets, float* __restrict__ out) {
  __shared__ float sA[256], sB[256];
  const int m = threadIdx.x;
  const float mx = rowmax[m], se = rowsum[m];
  const float lse = mx + logf(se);
  const int tgt = (int)targets[m];
  const float lt = logits[(size_t)m * NEM + tgt];
  const float L4 = -9.210340371976184f;     // log(1e-4)
  const float LH = -0.6931471805599453f;    // log(0.5)

  float beta  = -(lt - lse);
  float alpha = (expf(lt - mx) / se) * L4;
#pragma unroll
  for (int k = 0; k < 6; k++) {
    const int idx = t6i[m * 6 + k];
    if (idx != tgt && recip[m * 6 + k]) {
      const float v = t6v[m * 6 + k];
      beta  -= 0.5f * (v - lse);
      alpha += (expf(v - mx) / se) * (L4 - LH);
    }
  }
  alpha = -L4 + alpha;

  sA[m] = alpha; sB[m] = beta;
  __syncthreads();
  for (int off = 128; off > 0; off >>= 1) {
    if (m < off) { sA[m] += sA[m + off]; sB[m] += sB[m + off]; }
    __syncthreads();
  }
  if (m == 0) {
    out[0] = 0.05f * (sA[0] / 256.0f);  // UL_ALPHA * alpha_loss
    out[1] = 1.00f * (sB[0] / 256.0f);  // UL_BETA  * beta_loss
  }
}

// ===========================================================================
extern "C" void kernel_launch(void* const* d_in, const int* in_sizes, int n_in,
                              void* d_out, int out_size, void* d_ws,
                              size_t ws_size, hipStream_t stream) {
  const float*     inp = (const float*)d_in[0];
  const float*     em  = (const float*)d_in[1];
  const long long* tgt = (const long long*)d_in[2];
  float* out = (float*)d_out;
  char* ws = (char*)d_ws;

  size_t off = 0;
  float* logits = (float*)(ws + off); off += (size_t)BROWS * NEM * 4;
  float* t6v    = (float*)(ws + off); off += BROWS * 6 * 4;
  int*   t6i    = (int*)  (ws + off); off += BROWS * 6 * 4;
  float* rowmax = (float*)(ws + off); off += BROWS * 4;
  float* rowsum = (float*)(ws + off); off += BROWS * 4;
  int*   recip  = (int*)  (ws + off); off += BROWS * KNN6 * 4;
  float* pt6v   = (float*)(ws + off); off += (size_t)BROWS * KNN6 * NSPLIT * 6 * 4;
  int*   pt6i   = (int*)  (ws + off); off += (size_t)BROWS * KNN6 * NSPLIT * 6 * 4;

  // bf16 hi/lo planes (em + inputs): 2*(67.1 + 1.0) MB
  bf16* emhi  = (bf16*)(ws + off); off += EMTOT * 2;
  bf16* emlo  = (bf16*)(ws + off); off += EMTOT * 2;
  bf16* inphi = (bf16*)(ws + off); off += INTOT * 2;
  bf16* inplo = (bf16*)(ws + off); off += INTOT * 2;
  const bool planes = (ws_size >= off);

  if (planes) {
    const int nconv = (int)((EMTOT + INTOT) / 8 / 256);  // 16640
    k_convert <<<nconv, 256, 0, stream>>>(em, inp, emhi, emlo, inphi, inplo);
    k_logits_p<<<dim3(NEM / 128, BROWS / 64), 256, 0, stream>>>(inphi, inplo,
                                                                emhi, emlo, logits);
    k_rowstats<<<BROWS, 256, 0, stream>>>(logits, t6v, t6i, rowmax, rowsum);
    k_sims_p  <<<dim3((BROWS * KNN6) / 16, NSPLIT), 256, 0, stream>>>(
        emhi, emlo, t6i, pt6v, pt6i);
  } else {
    k_logits_f<<<dim3(NEM / 128, BROWS / 64), 256, 0, stream>>>(inp, em, logits);
    k_rowstats<<<BROWS, 256, 0, stream>>>(logits, t6v, t6i, rowmax, rowsum);
    k_sims_f  <<<dim3((BROWS * KNN6) / 16, NSPLIT), 256, 0, stream>>>(
        em, t6i, pt6v, pt6i);
  }
  k_recip<<<(BROWS * KNN6 + 255) / 256, 256, 0, stream>>>(pt6v, pt6i, t6i, recip);
  k_loss <<<1, 256, 0, stream>>>(logits, t6v, t6i, rowmax, rowsum, recip, tgt, out);
}
